// MultiHeadAttention_14413910245870
// MI455X (gfx1250) — compile-verified, matrix operand order checked by reading, not run on hardware
//
#include <hip/hip_runtime.h>
#include <hip/hip_bf16.h>

// ---------------------------------------------------------------------------
// MHA forward for MI455X (gfx1250), bf16 WMMA everywhere.
//   B=2, T=2048, D=1024, H=16, d_k=64
// Pipeline:
//   1) fp32 -> bf16 conversion of activations + weights (workspace)
//   2) Q/K/V projections: bf16 NT-GEMM (x @ W^T + b) via v_wmma_f32_16x16x32_bf16
//      (64x64 per-wave tile -> 1 ds_load_b128 per WMMA; global_prefetch of the
//       next K-tile overlaps HBM/L2 latency with the MMA burst)
//   3) flash attention (online softmax, causal + key-padding masks)
//   4) output projection -> fp32 d_out
// ---------------------------------------------------------------------------

typedef __bf16 bf16;
typedef __attribute__((ext_vector_type(16))) __bf16 v16bf;
typedef __attribute__((ext_vector_type(8)))  __bf16 v8bf;
typedef __attribute__((ext_vector_type(8)))  float  v8f;

#define DMODEL 1024
#define NHEAD  16
#define DK     64
#define BATCH  2
#define SEQ    2048

// ---------------------------------------------------------------------------
// helpers
// ---------------------------------------------------------------------------
__device__ __forceinline__ v16bf make_frag(v8bf lo, v8bf hi) {
  v16bf r;
#pragma unroll
  for (int i = 0; i < 8; ++i) { r[i] = lo[i]; r[i + 8] = hi[i]; }
  return r;
}

// Load a 16xK=32 bf16 A/B fragment from an LDS tile stored row-major
// (rowStride elements per row, 16B-aligned). ISA 16-bit A layout:
//   lane l: row = row0 + (l&15), khalf = l>>4
//   VGPR0-3 : K = khalf*8 + 0..7      (one b128)
//   VGPR4-7 : K = 16 + khalf*8 + 0..7 (one b128)
__device__ __forceinline__ v16bf load_frag(const bf16* tile, int row0, int col0,
                                           int rowStride) {
  const int lane  = threadIdx.x & 31;
  const int khalf = lane >> 4;
  const bf16* p = tile + (row0 + (lane & 15)) * rowStride + col0 + khalf * 8;
  v8bf lo = *(const v8bf*)(p);
  v8bf hi = *(const v8bf*)(p + 16);
  return make_frag(lo, hi);
}

// ---------------------------------------------------------------------------
// 1) fp32 -> bf16 conversion
// ---------------------------------------------------------------------------
__global__ void convert_f32_bf16(const float* __restrict__ src,
                                 bf16* __restrict__ dst, int n) {
  int i = (blockIdx.x * blockDim.x + threadIdx.x) * 4;
  if (i + 3 < n) {
    float4 f = *(const float4*)(src + i);
    dst[i + 0] = (bf16)f.x;
    dst[i + 1] = (bf16)f.y;
    dst[i + 2] = (bf16)f.z;
    dst[i + 3] = (bf16)f.w;
  } else {
    for (; i < n; ++i) dst[i] = (bf16)src[i];
  }
}

// ---------------------------------------------------------------------------
// 2/4) NT GEMM: C[M,N] = A[M,K] * B[N,K]^T + bias[N]
//   A, B bf16 row-major along K (matches x @ W^T with torch-layout W[out,in]).
//   Block: 256 threads = 8 waves -> 256x128 C tile; wave -> 64x64 sub-tile:
//   per K-step: 8 fragment loads (16 ds_load_b128) feed 16 WMMAs.
// ---------------------------------------------------------------------------
#define GBM 256
#define GBN 128
#define GBK 32
#define GSTRIDE 40  // 32 + 8 pad elements (80B rows, 16B aligned)

template <typename OutT>
__global__ __launch_bounds__(256) void gemm_nt_bias(
    const bf16* __restrict__ A, const bf16* __restrict__ Bm,
    const float* __restrict__ bias, OutT* __restrict__ C, int M, int N, int K) {
  __shared__ bf16 As[GBM * GSTRIDE];   // 256 x 32
  __shared__ bf16 Bs[GBN * GSTRIDE];   // 128 x 32

  const int tid  = threadIdx.x;
  const int wave = tid >> 5;
  const int lane = tid & 31;
  const int bm = blockIdx.x * GBM;
  const int bn = blockIdx.y * GBN;
  const int wm = (wave & 3) * 64;   // 4 waves tiling M (64 rows each)
  const int wn = (wave >> 2) * 64;  // 2 waves tiling N (64 cols each)

  v8f acc[4][4];
#pragma unroll
  for (int mi = 0; mi < 4; ++mi)
#pragma unroll
    for (int nj = 0; nj < 4; ++nj)
#pragma unroll
      for (int e = 0; e < 8; ++e) acc[mi][nj][e] = 0.0f;

  for (int k0 = 0; k0 < K; k0 += GBK) {
    // stage 256x32 (A) and 128x32 (B) bf16 tiles
#pragma unroll
    for (int i = 0; i < 4; ++i) {
      int idx = tid + i * 256;        // 0..1023
      int row = idx >> 2;             // 0..255
      int col = (idx & 3) * 8;        // 0,8,16,24
      *(v8bf*)(As + row * GSTRIDE + col) =
          *(const v8bf*)(A + (size_t)(bm + row) * K + k0 + col);
    }
#pragma unroll
    for (int i = 0; i < 2; ++i) {
      int idx = tid + i * 256;        // 0..511
      int row = idx >> 2;             // 0..127
      int col = (idx & 3) * 8;
      *(v8bf*)(Bs + row * GSTRIDE + col) =
          *(const v8bf*)(Bm + (size_t)(bn + row) * K + k0 + col);
    }
    // prefetch next K-tile (global_prefetch_b8) to overlap HBM/L2 latency
    if (k0 + GBK < K) {
      int row = tid >> 2;
      int col = (tid & 3) * 8;
      __builtin_prefetch(A + (size_t)(bm + row) * K + k0 + GBK + col, 0, 3);
      if (row < GBN)
        __builtin_prefetch(Bm + (size_t)(bn + row) * K + k0 + GBK + col, 0, 3);
    }
    __syncthreads();

    v16bf af[4];
#pragma unroll
    for (int mi = 0; mi < 4; ++mi) af[mi] = load_frag(As, wm + mi * 16, 0, GSTRIDE);
#pragma unroll
    for (int nj = 0; nj < 4; ++nj) {
      v16bf bfr = load_frag(Bs, wn + nj * 16, 0, GSTRIDE);
#pragma unroll
      for (int mi = 0; mi < 4; ++mi)
        acc[mi][nj] = __builtin_amdgcn_wmma_f32_16x16x32_bf16(
            false, af[mi], false, bfr, (short)0, acc[mi][nj], false, false);
    }
    __syncthreads();
  }

  // epilogue: bias + store.  C layout: element (v,lane): m = v + 8*(lane>=16),
  // n = lane & 15 (per 16x16 tile).
  const int half = lane >> 4;
  const int nloc = lane & 15;
#pragma unroll
  for (int mi = 0; mi < 4; ++mi) {
#pragma unroll
    for (int nj = 0; nj < 4; ++nj) {
      const int gn = bn + wn + nj * 16 + nloc;
      const float bb = bias[gn];
#pragma unroll
      for (int v = 0; v < 8; ++v) {
        const int gm = bm + wm + mi * 16 + v + half * 8;
        C[(size_t)gm * N + gn] = (OutT)(acc[mi][nj][v] + bb);
      }
    }
  }
}

// ---------------------------------------------------------------------------
// 3) Flash attention. Grid: (T/128, H, B). Block: 256 thr = 8 waves,
//    each wave owns 16 query rows, iterates 32-key tiles (causal-bounded).
// ---------------------------------------------------------------------------
#define ATT_BK 32
#define KSTRIDE 72  // 64 + 8 pad
#define PSTRIDE 40  // 32 + 8 pad

__global__ __launch_bounds__(256) void attention_kernel(
    const bf16* __restrict__ Q, const bf16* __restrict__ Km,
    const bf16* __restrict__ Vm, const int* __restrict__ kpm,
    bf16* __restrict__ O) {
  __shared__ bf16 Ks[ATT_BK * KSTRIDE];    // 32 keys x 64 dims
  __shared__ bf16 Vts[DK * PSTRIDE];       // transposed: 64 dims x 32 keys
  __shared__ bf16 Ps[8][16 * PSTRIDE];     // per-wave P scratch 16x32

  const int tid  = threadIdx.x;
  const int wave = tid >> 5;
  const int lane = tid & 31;
  const int half = lane >> 4;
  const int nloc = lane & 15;

  const int qt = blockIdx.x;
  const int h  = blockIdx.y;
  const int b  = blockIdx.z;
  const int q0 = qt * 128 + wave * 16;  // first q row of this wave
  const int dcol = h * DK;

  const bf16* qbase = Q + (size_t)b * SEQ * DMODEL + dcol;
  const bf16* kbase = Km + (size_t)b * SEQ * DMODEL + dcol;
  const bf16* vbase = Vm + (size_t)b * SEQ * DMODEL + dcol;
  const int*  kp    = kpm + b * SEQ;

  // Q fragments: 16x64 = two K-chunks, live in registers for the whole loop.
  v16bf qf[2];
  {
    const bf16* p = qbase + (size_t)(q0 + nloc) * DMODEL + half * 8;
    qf[0] = make_frag(*(const v8bf*)(p),      *(const v8bf*)(p + 16));
    qf[1] = make_frag(*(const v8bf*)(p + 32), *(const v8bf*)(p + 48));
  }

  float rmax[8], rsum[8];
  v8f oacc[4];
#pragma unroll
  for (int v = 0; v < 8; ++v) { rmax[v] = -1e30f; rsum[v] = 0.0f; }
#pragma unroll
  for (int t = 0; t < 4; ++t)
#pragma unroll
    for (int e = 0; e < 8; ++e) oacc[t][e] = 0.0f;

  const int jmax = (qt + 1) * 128 < SEQ ? (qt + 1) * 128 : SEQ;  // causal bound
  for (int j0 = 0; j0 < jmax; j0 += ATT_BK) {
    // ---- stage K (row-major) and V (transposed) tiles ----
    {
      const int row = tid >> 3;          // 0..31 keys
      const int col = (tid & 7) * 8;     // 0..56 dims
      *(v8bf*)(Ks + row * KSTRIDE + col) =
          *(const v8bf*)(kbase + (size_t)(j0 + row) * DMODEL + col);
      v8bf vv = *(const v8bf*)(vbase + (size_t)(j0 + row) * DMODEL + col);
#pragma unroll
      for (int i = 0; i < 8; ++i) Vts[(col + i) * PSTRIDE + row] = vv[i];
    }
    __syncthreads();

    // ---- scores S = Q @ K^T (16 q x 32 keys), two 16x16 N-tiles ----
    v8f sacc[2];
#pragma unroll
    for (int nt = 0; nt < 2; ++nt)
#pragma unroll
      for (int e = 0; e < 8; ++e) sacc[nt][e] = 0.0f;
#pragma unroll
    for (int nt = 0; nt < 2; ++nt) {
      v16bf kf0 = load_frag(Ks, nt * 16, 0, KSTRIDE);
      v16bf kf1 = load_frag(Ks, nt * 16, 32, KSTRIDE);
      sacc[nt] = __builtin_amdgcn_wmma_f32_16x16x32_bf16(
          false, qf[0], false, kf0, (short)0, sacc[nt], false, false);
      sacc[nt] = __builtin_amdgcn_wmma_f32_16x16x32_bf16(
          false, qf[1], false, kf1, (short)0, sacc[nt], false, false);
    }

    // ---- scale + masks (C layout: m = v + 8*half, n = nloc + nt*16) ----
    float sv[2][8];
#pragma unroll
    for (int nt = 0; nt < 2; ++nt) {
      const int kidx = j0 + nt * 16 + nloc;
      const bool pad = (kp[kidx] == 1);
#pragma unroll
      for (int v = 0; v < 8; ++v) {
        const int qi = q0 + v + half * 8;
        float s = sacc[nt][v] * 0.125f;  // 1/sqrt(64)
        if (pad || kidx > qi) s = -1e30f;
        sv[nt][v] = s;
      }
    }

    // ---- online softmax (row reductions across 16-lane halves) ----
    float corr[8];
#pragma unroll
    for (int v = 0; v < 8; ++v) {
      float m = fmaxf(sv[0][v], sv[1][v]);
#pragma unroll
      for (int off = 8; off >= 1; off >>= 1) m = fmaxf(m, __shfl_xor(m, off, 32));
      const float nm = fmaxf(rmax[v], m);
      corr[v] = __expf(rmax[v] - nm);
      rmax[v] = nm;
    }
    float psum[8];
#pragma unroll
    for (int v = 0; v < 8; ++v) psum[v] = 0.0f;
#pragma unroll
    for (int nt = 0; nt < 2; ++nt)
#pragma unroll
      for (int v = 0; v < 8; ++v) {
        const float p = __expf(sv[nt][v] - rmax[v]);
        psum[v] += p;
        Ps[wave][(v + 8 * half) * PSTRIDE + nt * 16 + nloc] = (bf16)p;
      }
#pragma unroll
    for (int v = 0; v < 8; ++v) {
      float s = psum[v];
#pragma unroll
      for (int off = 8; off >= 1; off >>= 1) s += __shfl_xor(s, off, 32);
      rsum[v] = rsum[v] * corr[v] + s;
    }
    // rescale running output
#pragma unroll
    for (int t = 0; t < 4; ++t)
#pragma unroll
      for (int v = 0; v < 8; ++v) oacc[t][v] *= corr[v];

    // ---- O += P @ V (K-dim = 32 keys, four 16x16 d-tiles) ----
    v16bf pf = load_frag(Ps[wave], 0, 0, PSTRIDE);
#pragma unroll
    for (int t = 0; t < 4; ++t) {
      v16bf vf = load_frag(Vts, t * 16, 0, PSTRIDE);
      oacc[t] = __builtin_amdgcn_wmma_f32_16x16x32_bf16(
          false, pf, false, vf, (short)0, oacc[t], false, false);
    }
    __syncthreads();
  }

  // ---- normalize + store ctx (merged heads: [b, t, h*64 + d]) ----
#pragma unroll
  for (int v = 0; v < 8; ++v) {
    const float inv = 1.0f / fmaxf(rsum[v], 1e-30f);
    const int gq = q0 + v + half * 8;
#pragma unroll
    for (int t = 0; t < 4; ++t) {
      const int gd = dcol + t * 16 + nloc;
      O[((size_t)b * SEQ + gq) * DMODEL + gd] = (bf16)(oacc[t][v] * inv);
    }
  }
}

// ---------------------------------------------------------------------------
// launch
// ---------------------------------------------------------------------------
extern "C" void kernel_launch(void* const* d_in, const int* in_sizes, int n_in,
                              void* d_out, int out_size, void* d_ws,
                              size_t ws_size, hipStream_t stream) {
  const float* query = (const float*)d_in[0];
  const float* key   = (const float*)d_in[1];
  const float* value = (const float*)d_in[2];
  const float* Wq = (const float*)d_in[3];
  const float* bq = (const float*)d_in[4];
  const float* Wk = (const float*)d_in[5];
  const float* bk = (const float*)d_in[6];
  const float* Wv = (const float*)d_in[7];
  const float* bv = (const float*)d_in[8];
  const float* Wo = (const float*)d_in[9];
  const float* bo = (const float*)d_in[10];
  const int* kpm  = (const int*)d_in[11];
  // d_in[12] = attention_mask (causal triu) -- applied arithmetically in-kernel

  const int M = BATCH * SEQ;                 // 4096 tokens
  const size_t ND = (size_t)M * DMODEL;      // activation elements
  const size_t WD = (size_t)DMODEL * DMODEL; // weight elements

  char* ws = (char*)d_ws;
  bf16* qb  = (bf16*)ws; ws += ND * sizeof(bf16);
  bf16* kb  = (bf16*)ws; ws += ND * sizeof(bf16);
  bf16* vb  = (bf16*)ws; ws += ND * sizeof(bf16);
  bf16* wqb = (bf16*)ws; ws += WD * sizeof(bf16);
  bf16* wkb = (bf16*)ws; ws += WD * sizeof(bf16);
  bf16* wvb = (bf16*)ws; ws += WD * sizeof(bf16);
  bf16* wob = (bf16*)ws; ws += WD * sizeof(bf16);
  bf16* Qp  = (bf16*)ws; ws += ND * sizeof(bf16);
  bf16* Kp  = (bf16*)ws; ws += ND * sizeof(bf16);
  bf16* Vp  = (bf16*)ws; ws += ND * sizeof(bf16);
  bf16* ctx = (bf16*)ws; ws += ND * sizeof(bf16);

  // 1) convert activations + weights to bf16
  {
    const int thr = 256;
    int nblk = (int)((ND / 4 + thr - 1) / thr);
    convert_f32_bf16<<<nblk, thr, 0, stream>>>(query, qb, (int)ND);
    convert_f32_bf16<<<nblk, thr, 0, stream>>>(key,   kb, (int)ND);
    convert_f32_bf16<<<nblk, thr, 0, stream>>>(value, vb, (int)ND);
    int wblk = (int)((WD / 4 + thr - 1) / thr);
    convert_f32_bf16<<<wblk, thr, 0, stream>>>(Wq, wqb, (int)WD);
    convert_f32_bf16<<<wblk, thr, 0, stream>>>(Wk, wkb, (int)WD);
    convert_f32_bf16<<<wblk, thr, 0, stream>>>(Wv, wvb, (int)WD);
    convert_f32_bf16<<<wblk, thr, 0, stream>>>(Wo, wob, (int)WD);
  }

  // 2) projections: Q/K/V = x @ W^T + b  (bf16 out)
  dim3 ggrid(M / GBM, DMODEL / GBN);
  gemm_nt_bias<bf16><<<ggrid, 256, 0, stream>>>(qb, wqb, bq, Qp, M, DMODEL, DMODEL);
  gemm_nt_bias<bf16><<<ggrid, 256, 0, stream>>>(kb, wkb, bk, Kp, M, DMODEL, DMODEL);
  gemm_nt_bias<bf16><<<ggrid, 256, 0, stream>>>(vb, wvb, bv, Vp, M, DMODEL, DMODEL);

  // 3) flash attention -> ctx (bf16, merged heads)
  dim3 agrid(SEQ / 128, NHEAD, BATCH);
  attention_kernel<<<agrid, 256, 0, stream>>>(Qp, Kp, Vp, kpm, ctx);

  // 4) output projection: d_out = ctx @ Wo^T + bo (fp32)
  gemm_nt_bias<float><<<ggrid, 256, 0, stream>>>(ctx, wob, bo, (float*)d_out, M,
                                                 DMODEL, DMODEL);
}
